// Block_82566451298336
// MI455X (gfx1250) — compile-verified, matrix operand order checked by reading, not run on hardware
//
#include <hip/hip_runtime.h>

// ---------------- types ----------------
typedef __attribute__((ext_vector_type(16))) __bf16        v16bf;
typedef __attribute__((ext_vector_type(8)))  float         v8f;
typedef __attribute__((ext_vector_type(4)))  unsigned int  v4u;
typedef __attribute__((ext_vector_type(4)))  unsigned int  u32x4;
typedef __attribute__((ext_vector_type(8)))  int           i32x8;
typedef __attribute__((ext_vector_type(4)))  int           i32x4;

// ---------------- problem constants ----------------
#define C_IN   256
#define C_OUT  256
#define H_IN   56
#define W_IN   56
#define H_OUT  54
#define W_OUT  54
#define NIMG   32
#define KDIM   2304                    // C_IN * 9 (OIHW flattens to [M][K] row-major)
#define PQ     (H_OUT * W_OUT)         // 2916
#define NPIX   (NIMG * PQ)             // 93312 GEMM columns
#define CHW    (C_IN * H_IN * W_IN)    // x image stride
#define HWIN   (H_IN * W_IN)           // x channel stride

// TDM availability (arity differs between toolchains)
#if defined(__has_builtin)
#if __has_builtin(__builtin_amdgcn_tensor_load_to_lds) && __has_builtin(__builtin_amdgcn_s_wait_tensorcnt)
#define HAVE_TDM 1
#endif
#endif
#ifndef HAVE_TDM
#define HAVE_TDM 0
#endif

// ---------------- prep: binarize weights (bf16) and bias (f32) ----------------
__global__ void prep_kernel(const float* __restrict__ w, const float* __restrict__ bias,
                            __bf16* __restrict__ wb, float* __restrict__ bb) {
    int i = blockIdx.x * blockDim.x + threadIdx.x;
    if (i < C_OUT * KDIM) wb[i] = (w[i] >= 0.0f) ? (__bf16)1.0f : (__bf16)-1.0f;
    if (i < C_OUT)        bb[i] = (bias[i] >= 0.0f) ? 1.0f : -1.0f;
}

// ---------------- conv as implicit GEMM via v_wmma_f32_16x16x32_bf16 ----------------
// block = 128 threads = 4 waves; tile = ALL 256 out-channels x 64 pixels
// (one block covers every channel -> im2col x bytes are staged once, not 4x)
// each wave: 64 channels x 64 pixels = 16 WMMA accumulators per K-step
// Double-buffered LDS + TDM: chunk i+1 DMA/staging overlaps chunk i compute,
// one barrier per iteration.  y layout: [C_OUT][NPIX] f32.
__global__ __launch_bounds__(128)
void conv_wmma_kernel(const float* __restrict__ x, const __bf16* __restrict__ wb,
                      const float* __restrict__ bb, float* __restrict__ y) {
    // 40-half (80 B) padded rows: 16B-aligned ds_load_b128 frags, conflict-free banks
#if HAVE_TDM
    __shared__ __align__(16) __bf16 ldsA[2][C_OUT][40]; // TDM dest (pad_amount -> 80B rows)
#endif
    __shared__ __align__(16) __bf16 ldsB[2][64][40];    // im2col: 64 pixels x 32 halves
    __shared__ int ldsOffs[KDIM];                       // k -> c*HWIN + r*W_IN + s

    const int tid    = threadIdx.x;
    const int lane   = tid & 31;
    const int wave   = tid >> 5;
    const int col    = lane & 15;       // row/col within a 16x16 fragment
    const int hiHalf = lane >> 4;       // 0: K 0-7/16-23,  1: K 8-15/24-31
    const int pixBase = blockIdx.x * 64;
    const int mWave   = wave * 64;      // this wave's 64-channel slice

    // ---- per-thread im2col base: pixel column is FIXED (tid & 63) ----
    const int nl   = tid & 63;
    const int klo  = tid >> 6;          // 0 or 1: interleave k-slots across thread halves
    {
        // offs table: one div chain per entry, once per block
        for (int t = tid; t < KDIM; t += 128) {
            int c  = t / 9;
            int rs = t - c * 9;
            int rr = rs / 3;
            int ss = rs - rr * 3;
            ldsOffs[t] = c * HWIN + rr * W_IN + ss;
        }
    }
    int pix = pixBase + nl;
    int img = pix / PQ;
    int rem = pix - img * PQ;
    int p   = rem / W_OUT;
    int q   = rem - p * W_OUT;
    const unsigned xBase = (unsigned)(img * CHW + p * W_IN + q);

    // ---- accumulators: 4 M-groups x 4 N-groups, init with binarized bias ----
    v8f acc[4][4];
    #pragma unroll
    for (int g = 0; g < 4; ++g) {
        #pragma unroll
        for (int r = 0; r < 8; ++r) {
            float b = bb[mWave + g * 16 + r + hiHalf * 8];
            acc[g][0][r] = b; acc[g][1][r] = b; acc[g][2][r] = b; acc[g][3][r] = b;
        }
    }

#if !HAVE_TDM
    const __bf16* arow[4];
    #pragma unroll
    for (int g = 0; g < 4; ++g)
        arow[g] = wb + (size_t)(mWave + g * 16 + col) * KDIM;
    const int kbase = hiHalf * 8;
#endif

#if HAVE_TDM
    // ---- TDM issue helper (wave 0 only; all operands block-uniform) ----
    auto issueTDM = [&](int kc, int b) {
        const __bf16* gA = wb + kc;                    // rows 0..255, cols kc..kc+31
        unsigned long long ga = (unsigned long long)(size_t)gA;
        unsigned ldsOff = (unsigned)(size_t)(void*)&ldsA[b][0][0];
        u32x4 g0;
        g0[0] = 1u;                                    // count=1 valid, user mode
        g0[1] = ldsOff;                                // lds_addr (bytes)
        g0[2] = (unsigned)(ga & 0xFFFFFFFFu);          // global_addr[31:0]
        g0[3] = (unsigned)((ga >> 32) & 0x1FFFFFFu)    // global_addr[56:32]
                | (2u << 30);                          // type = 2 (image)
        i32x8 g1;
        g1[0] = (int)((1u << 16)        // data_size = 2 bytes
                    | (1u << 20)        // pad_enable
                    | (3u << 22)        // pad_interval: 16 dwords (one 64B row)
                    | (3u << 25));      // pad_amount: 4 dwords -> 80B LDS row stride
        g1[1] = (int)((KDIM & 0xFFFFu) << 16);         // tensor_dim0[15:0]
        g1[2] = (int)(((KDIM >> 16) & 0xFFFFu)
                    | (4096u << 16));                  // tensor_dim1[15:0] (generous)
        g1[3] = (int)(32u << 16);                      // tile_dim0 = 32 elements
        g1[4] = C_OUT;                                 // tile_dim1 = 256 rows
        g1[5] = KDIM;                                  // tensor_dim0_stride[31:0]
        g1[6] = 0;
        g1[7] = 0;
        i32x4 gz = {0, 0, 0, 0};                       // groups 2/3 unused (2D tile)
#if __clang_major__ >= 23
        i32x8 gz8 = {0, 0, 0, 0, 0, 0, 0, 0};
        __builtin_amdgcn_tensor_load_to_lds(g0, g1, gz, gz, gz8, 0);
#else
        __builtin_amdgcn_tensor_load_to_lds(g0, g1, gz, gz, 0);
#endif
    };
#endif

    // ---- B staging: 32 K x 64 pixels; offs via broadcast ds_load (uniform per wave) ----
    auto stageB = [&](int kc, int b) {
        #pragma unroll
        for (int i = 0; i < 16; ++i) {
            int kl = 2 * i + klo;                      // 0..31
            unsigned idx = xBase + (unsigned)ldsOffs[kc + kl];
            ldsB[b][nl][kl] = (__bf16)x[idx];
        }
    };

    // ---- prologue: fill offs, then prepare chunk 0 in buffer 0 ----
#if HAVE_TDM
    if (wave == 0) issueTDM(0, 0);
#endif
    __syncthreads();                 // offs table visible to all stagers
    stageB(0, 0);
#if HAVE_TDM
    if (wave == 0) __builtin_amdgcn_s_wait_tensorcnt(0);
#endif
    __syncthreads();                 // chunk 0 ready

    int buf = 0;
    for (int kc = 0; kc < KDIM; kc += 32) {
        const int  nb      = buf ^ 1;
        const bool hasNext = (kc + 32 < KDIM);         // block-uniform

        if (hasNext) {
#if HAVE_TDM
            if (wave == 0) issueTDM(kc + 32, nb);      // DMA next A chunk async
#endif
            stageB(kc + 32, nb);                       // stage next B chunk
        }

        // ---- compute chunk `buf`: 4 A-frags x 4 B-frags = 16 WMMA ----
        union { v16bf v; v4u q[2]; } fa[4];
#if HAVE_TDM
        #pragma unroll
        for (int g = 0; g < 4; ++g) {
            const v4u* a4 = (const v4u*)(&ldsA[buf][mWave + g * 16 + col][0]);
            fa[g].q[0] = a4[hiHalf];
            fa[g].q[1] = a4[hiHalf + 2];
        }
#else
        #pragma unroll
        for (int g = 0; g < 4; ++g) {
            fa[g].q[0] = *(const v4u*)(arow[g] + kc + kbase);
            fa[g].q[1] = *(const v4u*)(arow[g] + kc + kbase + 16);
        }
#endif
        #pragma unroll
        for (int j = 0; j < 4; ++j) {
            union { v16bf v; v4u q[2]; } fb;
            const v4u* b4 = (const v4u*)(&ldsB[buf][j * 16 + col][0]);
            fb.q[0] = b4[hiHalf];
            fb.q[1] = b4[hiHalf + 2];
            #pragma unroll
            for (int g = 0; g < 4; ++g)
                acc[g][j] = __builtin_amdgcn_wmma_f32_16x16x32_bf16(
                                false, fa[g].v, false, fb.v,
                                (short)0, acc[g][j], false, false);
        }

#if HAVE_TDM
        if (hasNext && wave == 0) __builtin_amdgcn_s_wait_tensorcnt(0);
#endif
        __syncthreads();             // single barrier per iteration
        buf = nb;
    }

    // ---- store sixteen 16x16 f32 tiles of y ----
    #pragma unroll
    for (int g = 0; g < 4; ++g) {
        #pragma unroll
        for (int r = 0; r < 8; ++r) {
            int mC = mWave + g * 16 + r + hiHalf * 8;
            float* yr = y + (size_t)mC * NPIX + pixBase + col;
            yr[ 0] = acc[g][0][r];
            yr[16] = acc[g][1][r];
            yr[32] = acc[g][2][r];
            yr[48] = acc[g][3][r];
        }
    }
}

// ---------------- per-channel mean of y (variance not needed for sign) ----------------
__global__ __launch_bounds__(256)
void mean_kernel(const float* __restrict__ y, float* __restrict__ mean) {
    __shared__ float red[256];
    const float* row = y + (size_t)blockIdx.x * NPIX;
    float s = 0.0f;
    for (int i = threadIdx.x; i < NPIX; i += 256) s += row[i];
    red[threadIdx.x] = s;
    __syncthreads();
    for (int off = 128; off > 0; off >>= 1) {
        if (threadIdx.x < off) red[threadIdx.x] += red[threadIdx.x + off];
        __syncthreads();
    }
    if (threadIdx.x == 0) mean[blockIdx.x] = red[0] / (float)NPIX;
}

// ---------------- sign((y-mean)/std) == (y >= mean ? 1 : -1), remap to NCHW ----------------
__global__ void sign_kernel(const float* __restrict__ y, const float* __restrict__ mean,
                            float* __restrict__ out, int total) {
    int i = blockIdx.x * blockDim.x + threadIdx.x;
    if (i >= total) return;
    int n   = i / (C_OUT * PQ);
    int rem = i - n * (C_OUT * PQ);
    int k   = rem / PQ;
    int pq  = rem - k * PQ;
    float v = y[(size_t)k * NPIX + n * PQ + pq];
    out[i] = (v >= mean[k]) ? 1.0f : -1.0f;
}

// ---------------- launcher ----------------
extern "C" void kernel_launch(void* const* d_in, const int* in_sizes, int n_in,
                              void* d_out, int out_size, void* d_ws, size_t ws_size,
                              hipStream_t stream) {
    const float* x    = (const float*)d_in[0];   // (32,256,56,56)
    const float* w    = (const float*)d_in[1];   // (256,256,3,3) OIHW
    const float* bias = (const float*)d_in[2];   // (256,)
    float* out = (float*)d_out;                  // (32,256,54,54) f32

    // workspace layout (all 16B aligned)
    char*  ws = (char*)d_ws;
    float* y  = (float*)ws;                                   // 256*93312*4 = 95,551,488 B
    size_t yB = (size_t)C_OUT * NPIX * sizeof(float);
    __bf16* wb = (__bf16*)(ws + yB);                          // 256*2304*2 = 1,179,648 B
    size_t wbB = (size_t)C_OUT * KDIM * sizeof(__bf16);
    float* bb   = (float*)(ws + yB + wbB);                    // 256 f32
    float* mean = bb + C_OUT;                                 // 256 f32

    prep_kernel<<<(C_OUT * KDIM + 255) / 256, 256, 0, stream>>>(w, bias, wb, bb);

    conv_wmma_kernel<<<NPIX / 64, 128, 0, stream>>>(x, wb, bb, y);  // 1458 blocks

    mean_kernel<<<C_OUT, 256, 0, stream>>>(y, mean);

    int total = NIMG * C_OUT * PQ;
    sign_kernel<<<(total + 255) / 256, 256, 0, stream>>>(y, mean, out, total);
}
